// CumulantSOAP_CV_78176994722585
// MI455X (gfx1250) — matrix operand order repaired and checked by reading, not compiled, not run on hardware
//
#include <hip/hip_runtime.h>

typedef float     v4f  __attribute__((ext_vector_type(4)));
typedef float     v8f  __attribute__((ext_vector_type(8)));
typedef _Float16  v8h  __attribute__((ext_vector_type(8)));
typedef _Float16  v16h __attribute__((ext_vector_type(16)));

#define N_ROWS   200000
#define P        576
#define PC       (P * 3)        // 1728-element cumulant vector
#define NCHUNK   (PC / 32)      // 54 K=32 WMMA chunks
#define P4       144            // float4 columns (576/4)
#define CTILE    16             // float4-columns per block
#define RTHREADS 16             // row-lanes per block
#define ROWBLKS  120            // grid.y

// ---------------------------------------------------------------------------
// Kernel 0: zero the accumulation scratch (harness poisons d_ws, never resets)
// ---------------------------------------------------------------------------
__global__ void zero_ws(float* __restrict__ ws, int n) {
    int i = blockIdx.x * blockDim.x + threadIdx.x;
    if (i < n) ws[i] = 0.0f;
}

// ---------------------------------------------------------------------------
// Kernel 1: streaming column reduction (sum, sum of squares) over 460.8 MB.
// Pure single-pass stream: X exceeds the 192MB L2 and is used once, so
// nontemporal float4 loads (512B/wave32 access) + prefetch; per-thread
// register accumulation, LDS tree reduce, f32 global atomics into scratch.
// ---------------------------------------------------------------------------
__global__ __launch_bounds__(256)
void colreduce(const float* __restrict__ X,
               float* __restrict__ gsum,
               float* __restrict__ gsq) {
    const int tx   = threadIdx.x & (CTILE - 1);   // float4-column within tile
    const int ty   = threadIdx.x >> 4;            // row lane 0..15
    const int col4 = blockIdx.x * CTILE + tx;
    const v4f* __restrict__ X4 = (const v4f*)X;

    v4f s1 = {0.f, 0.f, 0.f, 0.f};
    v4f s2 = {0.f, 0.f, 0.f, 0.f};

    const int rstride = RTHREADS * gridDim.y;     // 1920 rows per step
    for (int r = blockIdx.y * RTHREADS + ty; r < N_ROWS; r += rstride) {
        const v4f* p = X4 + (size_t)r * P4 + col4;
        __builtin_prefetch((const void*)(p + (size_t)rstride * P4), 0, 0);
        v4f x = __builtin_nontemporal_load(p);    // streamed once: NT hint
        s1 += x;
        s2 += x * x;
    }

    __shared__ v4f sh1[RTHREADS][CTILE];
    __shared__ v4f sh2[RTHREADS][CTILE];
    sh1[ty][tx] = s1;
    sh2[ty][tx] = s2;
    __syncthreads();

    #pragma unroll
    for (int off = RTHREADS / 2; off > 0; off >>= 1) {
        if (ty < off) {
            sh1[ty][tx] += sh1[ty + off][tx];
            sh2[ty][tx] += sh2[ty + off][tx];
        }
        __syncthreads();
    }

    if (ty == 0) {
        v4f t1 = sh1[0][tx];
        v4f t2 = sh2[0][tx];
        #pragma unroll
        for (int c = 0; c < 4; ++c) {
            atomicAdd(&gsum[col4 * 4 + c], t1[c]);
            atomicAdd(&gsq [col4 * 4 + c], t2[c]);
        }
    }
}

// ---------------------------------------------------------------------------
// Kernel 2: finalize cumulants and project with WMMA (one wave).
//   cum[3p+0]=m, cum[3p+1]=0, cum[3p+2]=S2/N - m^2
//   out[n] = sum_j (cum[j]-mu[j]) * W[j][n],  n=0..3
// 54 chunks of K=32 via v_wmma_f32_16x16x32_f16, fully unrolled, two
// independent accumulators.
// A as packed here (contiguous k=16h+s, not AMD's table order): lanes {0,16} carry row M=0;
//   lane 0  halves: K={0..15}, lane 16 halves: K={16..31}.
//   -> per chunk: two contiguous 8-half (16B) runs at v_sh[k0+khalf*16] and +8.
//   Non-carrier lanes read a zero stub via pointer select (no exec branching).
// B as packed here (same contiguous k order as A): lane L = column N=L%16; lanes 0..15 K=0..15,
//   lanes 16..31 K=16..31 -> one contiguous 16-half run in column-major w_sh.
//   Lanes n>=4 read column n&3: harmless, since D[0,n] is per-column and we
//   only read D[0,0..3] (A rows m>0 are zero).
// D layout: VGPR0, lanes 0..15 = D[M=0][N=lane].
// ---------------------------------------------------------------------------
__global__ __launch_bounds__(32)
void finalize_wmma(const float* __restrict__ gsum,
                   const float* __restrict__ gsq,
                   const float* __restrict__ mu,
                   const float* __restrict__ W,
                   float* __restrict__ out) {
    __shared__ __attribute__((aligned(32))) _Float16 v_sh[PC];     // cum - mu
    __shared__ __attribute__((aligned(32))) _Float16 w_sh[4][PC];  // W col-major
    __shared__ __attribute__((aligned(32))) _Float16 zstub[32];    // zero source

    const int lane  = threadIdx.x;
    const int n     = lane & 15;
    const int khalf = lane >> 4;         // 0 or 1
    const float invN = 1.0f / (float)N_ROWS;

    zstub[lane] = (_Float16)0;
    for (int p = lane; p < P; p += 32) {
        float m    = gsum[p] * invN;
        float mom2 = gsq[p] * invN - m * m;
        v_sh[3 * p + 0] = (_Float16)(m    - mu[3 * p + 0]);
        v_sh[3 * p + 1] = (_Float16)(0.f  - mu[3 * p + 1]);
        v_sh[3 * p + 2] = (_Float16)(mom2 - mu[3 * p + 2]);
    }
    for (int k = lane; k < PC; k += 32) {
        #pragma unroll
        for (int nn = 0; nn < 4; ++nn)
            w_sh[nn][k] = (_Float16)W[k * 4 + nn];
    }
    __syncthreads();

    // Per-lane A source: lanes 0/16 walk v_sh, all others re-read the zero stub.
    const bool row0 = (n == 0);
    const _Float16* pA      = row0 ? (v_sh + khalf * 16) : zstub;
    const int       strideA = row0 ? 32 : 0;
    const _Float16* pB      = &w_sh[n & 3][khalf * 16];

    v8f acc0, acc1;
    #pragma unroll
    for (int i = 0; i < 8; ++i) { acc0[i] = 0.f; acc1[i] = 0.f; }

    #pragma unroll
    for (int c = 0; c < NCHUNK; c += 2) {
        // chunk c -> acc0
        {
            v8h lo = *(const v8h*)(pA + c * strideA);
            v8h hi = *(const v8h*)(pA + c * strideA + 8);
            v16h a;
            #pragma unroll
            for (int i = 0; i < 8; ++i) { a[i] = lo[i]; a[8 + i] = hi[i]; }
            v16h b = *(const v16h*)(pB + c * 32);
            acc0 = __builtin_amdgcn_wmma_f32_16x16x32_f16(
                false, a, false, b, (short)0, acc0, false, false);
        }
        // chunk c+1 -> acc1
        {
            v8h lo = *(const v8h*)(pA + (c + 1) * strideA);
            v8h hi = *(const v8h*)(pA + (c + 1) * strideA + 8);
            v16h a;
            #pragma unroll
            for (int i = 0; i < 8; ++i) { a[i] = lo[i]; a[8 + i] = hi[i]; }
            v16h b = *(const v16h*)(pB + (c + 1) * 32);
            acc1 = __builtin_amdgcn_wmma_f32_16x16x32_f16(
                false, a, false, b, (short)0, acc1, false, false);
        }
    }
    acc0 += acc1;

    if (lane < 4) out[lane] = acc0[0];   // D[0][n], lanes 0..3
}

// ---------------------------------------------------------------------------
extern "C" void kernel_launch(void* const* d_in, const int* in_sizes, int n_in,
                              void* d_out, int out_size, void* d_ws, size_t ws_size,
                              hipStream_t stream) {
    const float* X  = (const float*)d_in[0];   // (200000, 576) f32
    const float* mu = (const float*)d_in[1];   // (1728,) f32
    const float* W  = (const float*)d_in[2];   // (1728, 4) f32
    float* out  = (float*)d_out;               // (1, 4) f32

    float* gsum = (float*)d_ws;                // 576 floats
    float* gsq  = gsum + P;                    // 576 floats

    zero_ws<<<dim3((2 * P + 255) / 256), dim3(256), 0, stream>>>(gsum, 2 * P);

    dim3 grid(P4 / CTILE, ROWBLKS);            // 9 x 120 blocks
    colreduce<<<grid, dim3(256), 0, stream>>>(X, gsum, gsq);

    finalize_wmma<<<dim3(1), dim3(32), 0, stream>>>(gsum, gsq, mu, W, out);
}
